// ContinuousGRULayer_39530878992497
// MI455X (gfx1250) — matrix operand order repaired and checked by reading, not compiled, not run on hardware
//
#include <hip/hip_runtime.h>

typedef __attribute__((ext_vector_type(16))) _Float16 v16h;
typedef __attribute__((ext_vector_type(8)))  float    v8f;

#define Bsz 512
#define Tn  512
#define Dn  32
#define Hn  64
#define Ln  2
#define ALPHA 0.4f
#define BETA  0.8f

#define WAVES   4
#define THREADS (WAVES * 32)
#define NFRAG   84                      // 48 flow + 24 gru-Whh + 12 gru-Wih fragments
#define LDSW_HALVES (NFRAG * 512)       // 512 halves per fragment (32 lanes x 16)
#define LDSF_OFF    (LDSW_HALVES * 2)   // byte offset of float region
#define LDSF_FLOATS 1152                // flow bias 384 + wcol 384 + tw 128 + gru 256
#define LDS_BYTES   (LDSF_OFF + LDSF_FLOATS * 4)

// sigmoid(x) = rcp(1 + exp(-x))            (v_exp_f32 + v_rcp_f32, ~4 VALU)
__device__ __forceinline__ float sigm(float x) {
  return __builtin_amdgcn_rcpf(1.0f + __expf(-x));
}
// tanh(x) = (1 - e) * rcp(1 + e), e = exp(-2x)   (~5 VALU)
__device__ __forceinline__ float fast_tanh(float x) {
  float e = __expf(-2.0f * x);
  return (1.0f - e) * __builtin_amdgcn_rcpf(1.0f + e);
}

__device__ __forceinline__ v8f wmma16(v16h a, v16h b, v8f c) {
  // D = A(16x32 f16) * B(32x16 f16) + C(16x16 f32)
  return __builtin_amdgcn_wmma_f32_16x16x32_f16(false, a, false, b, (short)0, c, false, false);
}

// Load a pre-swizzled fp16 A-fragment from LDS (contiguous 32B per lane).
__device__ __forceinline__ v16h ldA(const _Float16* w, int frag, int lane) {
  return *(const v16h*)(w + frag * 512 + lane * 16);
}

__device__ __forceinline__ unsigned int pk2(float a, float b) {
  union { _Float16 h[2]; unsigned int u; } p;
  p.h[0] = (_Float16)a; p.h[1] = (_Float16)b;   // fuses into v_cvt_pk_f16_f32
  return p.u;
}

// Build fp16 B-fragment (K=32, N=16) from two f32 accumulator tiles (C/D layout).
// B layout: lo lanes hold K=0..15, hi lanes K=16..31, N = lane%16 (same N as C/D)
// -> pack to f16x2 dwords, then one half-wave swap (shfl_xor 16) + select.
__device__ __forceinline__ v16h bfrag(v8f q0, v8f q1, int hi) {
  union { v16h h; unsigned int d[8]; } u;
#pragma unroll
  for (int j = 0; j < 4; ++j) {
    unsigned int o0 = pk2(q0[2 * j], q0[2 * j + 1]);
    unsigned int o1 = pk2(q1[2 * j], q1[2 * j + 1]);
    unsigned int s0 = __shfl_xor(o0, 16, 32);
    unsigned int s1 = __shfl_xor(o1, 16, 32);
    u.d[j]     = hi ? s1 : o0;
    u.d[4 + j] = hi ? o1 : s0;
  }
  return u.h;
}

__global__ void __launch_bounds__(THREADS, 4)
cgru_scan_kernel(const float* __restrict__ x,    const float* __restrict__ tptr,
                 const float* __restrict__ Whr,  const float* __restrict__ bhr,
                 const float* __restrict__ Whz,  const float* __restrict__ bhz,
                 const float* __restrict__ Whh,  const float* __restrict__ bhhf,
                 const float* __restrict__ tw,
                 const float* __restrict__ Wih,  const float* __restrict__ Whhg,
                 const float* __restrict__ bih,  const float* __restrict__ bhhg,
                 float* __restrict__ out)
{
  extern __shared__ __align__(64) char smem[];
  _Float16* ldsW = (_Float16*)smem;
  float*    ldsF = (float*)(smem + LDSF_OFF);

  const int tid = threadIdx.x;
  const float* fW[3] = {Whr, Whz, Whh};
  const float* fB[3] = {bhr, bhz, bhhf};

  // ---- one-time: convert weights fp32->fp16, pre-swizzled into A-fragment layout ----
  // A-frag element (lane, j): m = lane%16, k_local = 16*(lane/16) + j
  for (int e = tid; e < NFRAG * 512; e += THREADS) {
    int frag = e >> 9, r = e & 511, lane = r >> 4, j = r & 15;
    int klocal = 16 * (lane >> 4) + j;
    int mrow = lane & 15;
    float v;
    if (frag < 48) {                 // flow: [l][gate][mt(4)][kt(2)]
      int l = frag / 24, rem = frag % 24, g = rem >> 3, f = rem & 7, mt = f >> 1, kt = f & 1;
      v = fW[g][(l * Hn + mt * 16 + mrow) * (Hn + 1) + kt * 32 + klocal];
    } else if (frag < 72) {          // gru W_hh [192,64]: [mt(12)][kt(2)]
      int f = frag - 48, mt = f >> 1, kt = f & 1;
      v = Whhg[(mt * 16 + mrow) * Hn + kt * 32 + klocal];
    } else {                         // gru W_ih [192,32]: [mt(12)]
      int mt = frag - 72;
      v = Wih[(mt * 16 + mrow) * Dn + klocal];
    }
    ldsW[e] = (_Float16)v;
  }
  // ---- float region: flow biases, W[:,64] cols (tt rank-1 term), tw, gru biases ----
  // gru region (256): [0,128): b_ih+b_hh presummed for r,z gates; [128,192): b_ih_n; [192,256): b_hh_n
  for (int i = tid; i < LDSF_FLOATS; i += THREADS) {
    float v;
    if (i < 384)       { int l = i / 192, g = (i % 192) / 64, m = i % 64; v = fB[g][l * Hn + m]; }
    else if (i < 768)  { int q = i - 384; int l = q / 192, g = (q % 192) / 64, m = q % 64;
                         v = fW[g][(l * Hn + m) * (Hn + 1) + Hn]; }
    else if (i < 896)  v = tw[i - 768];
    else {
      int q = i - 896;
      if (q < 128)      v = bih[q] + bhhg[q];     // r,z gates combined
      else if (q < 192) v = bih[q];               // b_ih n-gate (rows 128..191)
      else              v = bhhg[q - 64];         // b_hh n-gate (rows 128..191)
    }
    ldsF[i] = v;
  }
  __syncthreads();

  const int lane = tid & 31;
  const int wave = tid >> 5;
  const int nidx = lane & 15;                       // batch column within tile
  const int hi   = lane >> 4;
  const int bg   = (blockIdx.x * WAVES + wave) * 16 + nidx;  // this lane's batch row

  v8f h[4];
#pragma unroll
  for (int mt = 0; mt < 4; ++mt) h[mt] = v8f{};

  for (int t = 0; t < Tn; ++t) {
    float tt = tptr[(size_t)bg * Tn + t];

    // ================= GRU-flow (2 layers) =================
#pragma unroll
    for (int l = 0; l < Ln; ++l) {
      v16h hf[2];
#pragma unroll
      for (int kt = 0; kt < 2; ++kt) hf[kt] = bfrag(h[2 * kt], h[2 * kt + 1], hi);

      // pass 1: r gate only -> rh = r*h
      v8f rh[4];
#pragma unroll
      for (int mt = 0; mt < 4; ++mt) {
        v8f cr = v8f{};
#pragma unroll
        for (int kt = 0; kt < 2; ++kt)
          cr = wmma16(ldA(ldsW, (l * 3 + 0) * 8 + mt * 2 + kt, lane), hf[kt], cr);
        const float* br = ldsF +       (l * 3 + 0) * 64 + mt * 16 + hi * 8;
        const float* wr = ldsF + 384 + (l * 3 + 0) * 64 + mt * 16 + hi * 8;
#pragma unroll
        for (int v2 = 0; v2 < 8; ++v2)
          rh[mt][v2] = (BETA * sigm(cr[v2] + br[v2] + wr[v2] * tt)) * h[mt][v2];
      }
      v16h rhf[2];
#pragma unroll
      for (int kt = 0; kt < 2; ++kt) rhf[kt] = bfrag(rh[2 * kt], rh[2 * kt + 1], hi);

      // pass 2: z and u GEMMs interleaved per mt, h updated in place
#pragma unroll
      for (int mt = 0; mt < 4; ++mt) {
        v8f cz = v8f{}, cu = v8f{};
#pragma unroll
        for (int kt = 0; kt < 2; ++kt) {
          cz = wmma16(ldA(ldsW, (l * 3 + 1) * 8 + mt * 2 + kt, lane), hf[kt],  cz);
          cu = wmma16(ldA(ldsW, (l * 3 + 2) * 8 + mt * 2 + kt, lane), rhf[kt], cu);
        }
        const float* bz  = ldsF +       (l * 3 + 1) * 64 + mt * 16 + hi * 8;
        const float* wz  = ldsF + 384 + (l * 3 + 1) * 64 + mt * 16 + hi * 8;
        const float* bu  = ldsF +       (l * 3 + 2) * 64 + mt * 16 + hi * 8;
        const float* wu  = ldsF + 384 + (l * 3 + 2) * 64 + mt * 16 + hi * 8;
        const float* twp = ldsF + 768 + l * 64           + mt * 16 + hi * 8;
#pragma unroll
        for (int v2 = 0; v2 < 8; ++v2) {
          float z = ALPHA * sigm(cz[v2] + bz[v2] + wz[v2] * tt);
          float u = fast_tanh(cu[v2] + bu[v2] + wu[v2] * tt);
          float g = fast_tanh(twp[v2] * tt);
          h[mt][v2] = h[mt][v2] + g * (z * (u - h[mt][v2]));
        }
      }
    }

    // ---- store post-flow hidden: out[b, t, :] (8 contiguous floats per lane per tile) ----
    {
      float* op = out + ((size_t)bg * Tn + t) * Hn + hi * 8;
#pragma unroll
      for (int mt = 0; mt < 4; ++mt)
#pragma unroll
        for (int v2 = 0; v2 < 8; ++v2) op[mt * 16 + v2] = h[mt][v2];
    }

    // ================= GRU cell =================
    v16h hgf[2];
#pragma unroll
    for (int kt = 0; kt < 2; ++kt) hgf[kt] = bfrag(h[2 * kt], h[2 * kt + 1], hi);

    // x^T B-fragment straight from global: lane loads 16 contiguous floats of x[b,t,:]
    const float* xp = x + ((size_t)bg * Tn + t) * Dn + hi * 16;
    v16h xf;
#pragma unroll
    for (int j = 0; j < 16; ++j) xf[j] = (_Float16)xp[j];
    if (t + 1 < Tn) __builtin_prefetch(xp + Dn, 0, 1);   // global_prefetch_b8 for next step

#pragma unroll
    for (int j4 = 0; j4 < 4; ++j4) {
      // r,z: input-GEMM and hidden-GEMM fused into a single accumulator chain
      v8f cr = wmma16(ldA(ldsW, 72 + 0 + j4, lane), xf, v8f{});
      v8f cz = wmma16(ldA(ldsW, 72 + 4 + j4, lane), xf, v8f{});
      v8f gin = wmma16(ldA(ldsW, 72 + 8 + j4, lane), xf, v8f{});
      v8f ghn = v8f{};
#pragma unroll
      for (int kt = 0; kt < 2; ++kt) {
        cr  = wmma16(ldA(ldsW, 48 + (0 + j4) * 2 + kt, lane), hgf[kt], cr);
        cz  = wmma16(ldA(ldsW, 48 + (4 + j4) * 2 + kt, lane), hgf[kt], cz);
        ghn = wmma16(ldA(ldsW, 48 + (8 + j4) * 2 + kt, lane), hgf[kt], ghn);
      }
      const float* bsr = ldsF + 896 + 0   + j4 * 16 + hi * 8;   // b_ih_r + b_hh_r
      const float* bsz = ldsF + 896 + 64  + j4 * 16 + hi * 8;   // b_ih_z + b_hh_z
      const float* bin = ldsF + 896 + 128 + j4 * 16 + hi * 8;   // b_ih_n
      const float* bhn = ldsF + 896 + 192 + j4 * 16 + hi * 8;   // b_hh_n
#pragma unroll
      for (int v2 = 0; v2 < 8; ++v2) {
        float r  = sigm(cr[v2] + bsr[v2]);
        float z  = sigm(cz[v2] + bsz[v2]);
        float nn = fast_tanh(gin[v2] + bin[v2] + r * (ghn[v2] + bhn[v2]));
        h[j4][v2] = (1.0f - z) * nn + z * h[j4][v2];   // in-place: later j4 only reads hgf
      }
    }
  }
}

extern "C" void kernel_launch(void* const* d_in, const int* in_sizes, int n_in,
                              void* d_out, int out_size, void* d_ws, size_t ws_size,
                              hipStream_t stream) {
  (void)in_sizes; (void)n_in; (void)out_size; (void)d_ws; (void)ws_size;
  const float* x    = (const float*)d_in[0];
  const float* t    = (const float*)d_in[1];
  const float* Whr  = (const float*)d_in[2];
  const float* bhr  = (const float*)d_in[3];
  const float* Whz  = (const float*)d_in[4];
  const float* bhz  = (const float*)d_in[5];
  const float* Whh  = (const float*)d_in[6];
  const float* bhh  = (const float*)d_in[7];
  const float* tw   = (const float*)d_in[8];
  const float* Wih  = (const float*)d_in[9];
  const float* Whhg = (const float*)d_in[10];
  const float* bih  = (const float*)d_in[11];
  const float* bhhg = (const float*)d_in[12];
  float* out = (float*)d_out;

  hipFuncSetAttribute(reinterpret_cast<const void*>(cgru_scan_kernel),
                      hipFuncAttributeMaxDynamicSharedMemorySize, LDS_BYTES);
  dim3 grid(Bsz / (16 * WAVES));   // 8 blocks x 4 waves = 32 waves, 16 batch rows each
  cgru_scan_kernel<<<grid, THREADS, LDS_BYTES, stream>>>(
      x, t, Whr, bhr, Whz, bhz, Whh, bhh, tw, Wih, Whhg, bih, bhhg, out);
}
